// SparsePairwiseRelationModule_50251117363747
// MI455X (gfx1250) — compile-verified, matrix operand order checked by reading, not run on hardware
//
#include <hip/hip_runtime.h>
#include <hip/hip_bf16.h>

typedef __attribute__((ext_vector_type(16))) __bf16 v16bf;
typedef __attribute__((ext_vector_type(8)))  __bf16 v8bf;
typedef __attribute__((ext_vector_type(8)))  float  v8f;

#define B_    4
#define N_    2048
#define D_    320
#define L_    256
#define H_    256
#define P_    902          // 2*D + 6 + L
#define KNB   16           // top-k
#define KC_   29           // ceil(902/32) K-chunks of 32
#define KPAD  (KC_ * 32)   // 928
#define NT_   (H_ / 16)    // 16 column tiles of H
#define Q_    4            // queries per block (W1 L2-traffic amortization)

// ---------------------------------------------------------------------------
// Kernel 0: pre-swizzle W1 (f32 [902,256] row-major) into bf16 WMMA-B
// fragment order, zero-padded to K=928. Fragment (nt,kc) = 32x16 B tile;
// lane l owns 16 contiguous bf16 (one 32B load at runtime).
// ---------------------------------------------------------------------------
__global__ void prep_w1_kernel(const float* __restrict__ W1,
                               __bf16* __restrict__ w1p) {
  int e = blockIdx.x * blockDim.x + threadIdx.x;   // < 16*29*512
  int j    = e & 15;
  int l    = (e >> 4) & 31;
  int frag = e >> 9;
  int kc   = frag % KC_;
  int nt   = frag / KC_;
  int nl = l & 15, half = l >> 4;
  int r = j >> 1, p = j & 1;
  int klocal = (r < 4) ? (half * 8 + 2 * r + p)
                       : (16 + half * 8 + 2 * (r - 4) + p);
  int k = kc * 32 + klocal;
  int n = nt * 16 + nl;
  float v = (k < P_) ? W1[(size_t)k * H_ + n] : 0.0f;
  w1p[e] = (__bf16)v;
}

// ---------------------------------------------------------------------------
// Kernel 1: top-16 nearest neighbors. One thread per query row, centers
// tiled through LDS, 16-entry register insertion sort (stable like top_k).
// ---------------------------------------------------------------------------
__global__ __launch_bounds__(256)
void topk_kernel(const float* __restrict__ centers,
                 const unsigned char* __restrict__ mask,
                 int* __restrict__ wsIdx,
                 float* __restrict__ outIdx) {
  __shared__ float scx[256], scy[256], scz[256];
  __shared__ unsigned char sm[256];

  const int b = blockIdx.y;
  const int i = blockIdx.x * 256 + threadIdx.x;
  const float* cb = centers + (size_t)b * N_ * 3;

  const float cix = cb[i * 3 + 0];
  const float ciy = cb[i * 3 + 1];
  const float ciz = cb[i * 3 + 2];
  const bool  mi  = mask[(size_t)b * N_ + i] != 0;

  float dist[KNB];
  int   ind[KNB];
#pragma unroll
  for (int s = 0; s < KNB; ++s) { dist[s] = __builtin_inff(); ind[s] = s; }

  for (int tile = 0; tile < N_; tile += 256) {
    __syncthreads();
    const int j0 = tile + threadIdx.x;
    scx[threadIdx.x] = cb[j0 * 3 + 0];
    scy[threadIdx.x] = cb[j0 * 3 + 1];
    scz[threadIdx.x] = cb[j0 * 3 + 2];
    sm[threadIdx.x]  = mask[(size_t)b * N_ + j0];
    __syncthreads();
    for (int jj = 0; jj < 256; ++jj) {
      const int j = tile + jj;
      const float dx = cix - scx[jj];
      const float dy = ciy - scy[jj];
      const float dz = ciz - scz[jj];
      const float d  = (dx * dx + dy * dy + dz * dz) * (1.0f / 25.0f);
      const bool ok = mi && (sm[jj] != 0) && (j != i);
      if (ok && d < dist[KNB - 1]) {
        dist[KNB - 1] = d; ind[KNB - 1] = j;
#pragma unroll
        for (int s = KNB - 1; s > 0; --s) {
          if (dist[s] < dist[s - 1]) {
            float td = dist[s]; dist[s] = dist[s - 1]; dist[s - 1] = td;
            int   ti = ind[s];  ind[s]  = ind[s - 1];  ind[s - 1]  = ti;
          }
        }
      }
    }
  }
  const size_t base = ((size_t)b * N_ + i) * KNB;
#pragma unroll
  for (int s = 0; s < KNB; ++s) {
    wsIdx[base + s]  = ind[s];
    outIdx[base + s] = (float)ind[s];
  }
}

// ---------------------------------------------------------------------------
// Kernel 2: fused pair-MLP + softmax + context. One block (4 waves, 128
// threads) handles Q_=4 queries: four 16x928 bf16 A-tiles in LDS; each wave
// owns 4 H column tiles; every B fragment from L2 feeds 4 WMMAs (one per
// query) -> 4x less W1 L2 traffic.
// ---------------------------------------------------------------------------
__global__ __launch_bounds__(128)
void pair_mlp_kernel(const float* __restrict__ objf,
                     const float* __restrict__ lang,
                     const float* __restrict__ centers,
                     const float* __restrict__ sizes,
                     const __bf16* __restrict__ w1p,
                     const float* __restrict__ b1,
                     const float* __restrict__ W2,
                     const float* __restrict__ b2,
                     const unsigned char* __restrict__ mask,
                     const int* __restrict__ wsIdx,
                     float* __restrict__ outEnh,
                     float* __restrict__ outW) {
  __shared__ __attribute__((aligned(32))) __bf16 aLds[Q_ * KNB * KPAD]; // 118,784 B
  __shared__ float s_fi[Q_][D_];          // 5,120 B
  __shared__ float s_lang[L_];            // 1,024 B
  __shared__ float s_geo[Q_][KNB][6];     // 1,536 B
  __shared__ float s_score[Q_][KNB];
  __shared__ float s_w[Q_][KNB];
  __shared__ int   s_idx[Q_][KNB];

  const int    tid = threadIdx.x;
  const int    n0  = blockIdx.x * Q_;
  const int    b   = blockIdx.y;
  const size_t rowb = (size_t)b * N_;

  // ---- stage inputs -------------------------------------------------------
  if (tid < Q_ * KNB) {
    const int q = tid >> 4, m = tid & 15;
    const int j = wsIdx[(rowb + n0 + q) * KNB + m];
    s_idx[q][m]   = j;
    s_score[q][m] = 0.0f;
    const size_t ci = (rowb + n0 + q) * 3;
    const size_t cj = (rowb + j) * 3;
#pragma unroll
    for (int a = 0; a < 3; ++a) {
      s_geo[q][m][a]     = (centers[ci + a] - centers[cj + a]) * (1.0f / 5.000001f);
      s_geo[q][m][3 + a] = (sizes[ci + a]   - sizes[cj + a])   * 0.5f;
    }
  }
  for (int e = tid; e < L_; e += 128) s_lang[e] = lang[(size_t)b * L_ + e];
  for (int e = tid; e < Q_ * D_; e += 128) {
    const int q = e / D_, d = e - q * D_;
    s_fi[q][d] = objf[(rowb + n0 + q) * D_ + d];
  }
  __syncthreads();

  // ---- build Q_ A-tiles: [feat_i | neighbor_feat | rel_geom | lang | pad] --
  for (int e = tid; e < Q_ * KNB * KPAD; e += 128) {
    const int row = e / KPAD;      // q*16 + m
    const int c   = e - row * KPAD;
    const int q = row >> 4, m = row & 15;
    float v;
    if (c < D_)               v = s_fi[q][c];
    else if (c < 2 * D_)      v = objf[(rowb + s_idx[q][m]) * D_ + (c - D_)];
    else if (c < 2 * D_ + 6)  v = s_geo[q][m][c - 2 * D_];
    else if (c < P_)          v = s_lang[c - (2 * D_ + 6)];
    else                      v = 0.0f;
    aLds[e] = (__bf16)v;
  }
  __syncthreads();

  // ---- WMMA main loop -----------------------------------------------------
  const int lane = tid & 31;
  const int wv   = tid >> 5;
  const int half = lane >> 4;
  const int nl   = lane & 15;

  v8f acc[Q_][4] = {};
  for (int kc = 0; kc < KC_; ++kc) {
    v16bf afrag[Q_];
#pragma unroll
    for (int q = 0; q < Q_; ++q) {
      const __bf16* ap = &aLds[(q * KNB + nl) * KPAD + kc * 32 + half * 8];
      v8bf lo = *(const v8bf*)ap;
      v8bf hi = *(const v8bf*)(ap + 16);
      afrag[q] = __builtin_shufflevector(lo, hi,
          0, 1, 2, 3, 4, 5, 6, 7, 8, 9, 10, 11, 12, 13, 14, 15);
    }
#pragma unroll
    for (int t = 0; t < 4; ++t) {
      const int nt = wv * 4 + t;
      v16bf bfrag = *(const v16bf*)(w1p + (((size_t)(nt * KC_ + kc)) << 9) + lane * 16);
#pragma unroll
      for (int q = 0; q < Q_; ++q) {
        acc[q][t] = __builtin_amdgcn_wmma_f32_16x16x32_bf16(
            false, afrag[q], false, bfrag, (short)0, acc[q][t], false, false);
      }
    }
  }

  // ---- hidden = relu(acc + b1); score_m += sum_n hidden * W2[n] -----------
#pragma unroll
  for (int t = 0; t < 4; ++t) {
    const int col = (wv * 4 + t) * 16 + nl;
    const float bb  = b1[col];
    const float w2v = W2[col];
#pragma unroll
    for (int q = 0; q < Q_; ++q) {
#pragma unroll
      for (int v = 0; v < 8; ++v) {
        float h = fmaxf(acc[q][t][v] + bb, 0.0f);
        float p = h * w2v;
        p += __shfl_xor(p, 1, 32);
        p += __shfl_xor(p, 2, 32);
        p += __shfl_xor(p, 4, 32);
        p += __shfl_xor(p, 8, 32);        // sum over 16 lanes of each half
        if (nl == 0) atomicAdd(&s_score[q][v + 8 * half], p);
      }
    }
  }
  __syncthreads();

  // ---- softmax over 16 neighbor scores per query (tid<64 = 4 lane-groups) -
  if (tid < Q_ * KNB) {
    const int q = tid >> 4, m = tid & 15;
    float sc = s_score[q][m] + b2[0];     // TEMPERATURE == 1
    if (mask[rowb + s_idx[q][m]] == 0) sc = -1e9f;
    float mx = sc;
    mx = fmaxf(mx, __shfl_xor(mx, 1, 16));
    mx = fmaxf(mx, __shfl_xor(mx, 2, 16));
    mx = fmaxf(mx, __shfl_xor(mx, 4, 16));
    mx = fmaxf(mx, __shfl_xor(mx, 8, 16));
    const float ev = __expf(sc - mx);
    float ssum = ev;
    ssum += __shfl_xor(ssum, 1, 16);
    ssum += __shfl_xor(ssum, 2, 16);
    ssum += __shfl_xor(ssum, 4, 16);
    ssum += __shfl_xor(ssum, 8, 16);
    const float w = ev / ssum;
    s_w[q][m] = w;
    outW[(rowb + n0 + q) * KNB + m] = w;
  }
  __syncthreads();

  // ---- context + residual (f32 neighbor rows re-read; L2-resident) --------
  for (int e = tid; e < Q_ * D_; e += 128) {
    const int q = e / D_, d = e - q * D_;
    float ctx = 0.0f;
#pragma unroll
    for (int kk = 0; kk < KNB; ++kk)
      ctx += s_w[q][kk] * objf[(rowb + s_idx[q][kk]) * D_ + d];
    outEnh[(rowb + n0 + q) * D_ + d] = s_fi[q][d] + ctx;
  }
}

// ---------------------------------------------------------------------------
extern "C" void kernel_launch(void* const* d_in, const int* in_sizes, int n_in,
                              void* d_out, int out_size, void* d_ws, size_t ws_size,
                              hipStream_t stream) {
  const float* objf  = (const float*)d_in[0];
  const float* lang  = (const float*)d_in[1];
  const float* cents = (const float*)d_in[2];
  const float* szs   = (const float*)d_in[3];
  const float* W1    = (const float*)d_in[4];
  const float* b1    = (const float*)d_in[5];
  const float* W2    = (const float*)d_in[6];
  const float* b2    = (const float*)d_in[7];
  const unsigned char* om = (const unsigned char*)d_in[8];

  // workspace: [0, 512KB) swizzled bf16 W1 (475,136 B); then int indices
  __bf16* w1p  = (__bf16*)d_ws;
  int*   wsIdx = (int*)((char*)d_ws + (size_t)512 * 1024);

  float* outEnh = (float*)d_out;
  float* outW   = outEnh + (size_t)B_ * N_ * D_;
  float* outIdx = outW   + (size_t)B_ * N_ * KNB;

  prep_w1_kernel<<<dim3((NT_ * KC_ * 512) / 256), 256, 0, stream>>>(W1, w1p);
  topk_kernel<<<dim3(N_ / 256, B_), 256, 0, stream>>>(cents, om, wsIdx, outIdx);
  pair_mlp_kernel<<<dim3(N_ / Q_, B_), 128, 0, stream>>>(
      objf, lang, cents, szs, w1p, b1, W2, b2, om, wsIdx, outEnh, outW);
}
